// CustomGRUCell_26027501814359
// MI455X (gfx1250) — matrix operand order repaired and checked by reading, not compiled, not run on hardware
//
#include <hip/hip_runtime.h>
#include <stdint.h>

#define HH 2048
#define BB 4096
#define BM 128
#define BN 64
#define BK 32
#define BKP 40      // padded LDS row (halves) to spread banks
#define NTHREADS 256
#define KSTEPS (HH / BK)

// per-buffer strides in bytes for the double-buffered LDS tiles
#define ABUF_BYTES (2 * BM * BKP * 2)   // 20480
#define WBUF_BYTES (6 * BN * BKP * 2)   // 30720

typedef __attribute__((ext_vector_type(16))) __bf16    v16bf;
typedef __attribute__((ext_vector_type(8)))  float     v8f;
typedef __attribute__((ext_vector_type(4)))  unsigned  v4u;

union BFrag { v4u d[2]; v16bf v; };

__device__ __forceinline__ unsigned short f2bf(float f) {
    union { float f; unsigned u; } c; c.f = f;
    unsigned u = c.u;
    u += 0x7FFFu + ((u >> 16) & 1u);   // round-to-nearest-even
    return (unsigned short)(u >> 16);
}

// CDNA5 async DMA: copy 16B global -> LDS, tracked by ASYNCcnt (no VGPR data path)
__device__ __forceinline__ void async_copy16(unsigned ldsOff, const unsigned short* gptr) {
    asm volatile("global_load_async_to_lds_b128 %0, %1, off"
                 :: "v"(ldsOff), "v"(gptr)
                 : "memory");
}
__device__ __forceinline__ void wait_async0() {
    asm volatile("s_wait_asynccnt 0x0" ::: "memory");
}

// ---------------------------------------------------------------------------
// Kernel 1: fp32 -> bf16 for input and hx (layout preserved, vectorized)
// ---------------------------------------------------------------------------
__global__ void cvt_act(const float4* __restrict__ x, const float4* __restrict__ h,
                        unsigned short* __restrict__ xb, unsigned short* __restrict__ hb) {
    size_t i = (size_t)blockIdx.x * blockDim.x + threadIdx.x;   // one float4 per tensor
    float4 v = x[i];
    uint2 p;
    p.x = (unsigned)f2bf(v.x) | ((unsigned)f2bf(v.y) << 16);
    p.y = (unsigned)f2bf(v.z) | ((unsigned)f2bf(v.w) << 16);
    *(uint2*)(xb + i * 4) = p;
    v = h[i];
    uint2 q;
    q.x = (unsigned)f2bf(v.x) | ((unsigned)f2bf(v.y) << 16);
    q.y = (unsigned)f2bf(v.z) | ((unsigned)f2bf(v.w) << 16);
    *(uint2*)(hb + i * 4) = q;
}

// ---------------------------------------------------------------------------
// Kernel 2: weights fp32 [g][K][N] -> bf16 transposed [g][N][K] (tile transpose)
// ---------------------------------------------------------------------------
__global__ void cvt_wt(const float* __restrict__ wih, const float* __restrict__ whh,
                       unsigned short* __restrict__ wihT, unsigned short* __restrict__ whhT) {
    __shared__ unsigned short t[64][68];
    const int z = blockIdx.z;                 // 0..2 = ih gates, 3..5 = hh gates
    const float* src    = (z < 3) ? (wih  + (size_t)z       * HH * HH)
                                  : (whh  + (size_t)(z - 3) * HH * HH);
    unsigned short* dst = (z < 3) ? (wihT + (size_t)z       * HH * HH)
                                  : (whhT + (size_t)(z - 3) * HH * HH);
    const int kt = blockIdx.y * 64;
    const int nt = blockIdx.x * 64;
    const int r0 = threadIdx.x >> 4;          // 0..15
    const int c4 = (threadIdx.x & 15) * 4;    // 0,4,...,60

    #pragma unroll
    for (int rr = 0; rr < 4; ++rr) {
        int k = r0 + rr * 16;
        float4 v = *(const float4*)(src + (size_t)(kt + k) * HH + nt + c4);
        t[c4 + 0][k] = f2bf(v.x);
        t[c4 + 1][k] = f2bf(v.y);
        t[c4 + 2][k] = f2bf(v.z);
        t[c4 + 3][k] = f2bf(v.w);
    }
    __syncthreads();
    #pragma unroll
    for (int rr = 0; rr < 4; ++rr) {
        int n = r0 + rr * 16;
        uint2 w = *(const uint2*)&t[n][c4];
        *(uint2*)(dst + (size_t)(nt + n) * HH + kt + c4) = w;
    }
}

// ---------------------------------------------------------------------------
// Kernel 3: fused 6-way GEMM (bf16 WMMA, f32 accum) + GRU gate math.
// Double-buffered LDS, filled by CDNA5 async global->LDS DMA (ASYNCcnt),
// one barrier per k-step; 24 WMMAs per wave per k-step.
// ---------------------------------------------------------------------------
__global__ __launch_bounds__(NTHREADS) void gru_wmma(
    const unsigned short* __restrict__ Xb,   const unsigned short* __restrict__ Hb,
    const unsigned short* __restrict__ WihT, const unsigned short* __restrict__ WhhT,
    const float* __restrict__ bih, const float* __restrict__ bhh,
    const float* __restrict__ hx,  float* __restrict__ out) {

    __shared__ unsigned short As[2][2][BM][BKP];   // [buf][src][m][k]   40 KB
    __shared__ unsigned short Bs[2][6][BN][BKP];   // [buf][gate][n][k]  60 KB

    const int tid  = threadIdx.x;
    const int lane = tid & 31;
    const int lh   = lane >> 4;        // half-wave: 0 or 1
    const int lr   = lane & 15;
    const int wave = tid >> 5;         // 0..7
    const int wm   = wave >> 1;        // 0..3 -> M offset wm*32
    const int wn   = wave & 1;         // 0..1 -> N offset wn*32

    const int m0 = blockIdx.y * BM;
    const int n0 = blockIdx.x * BN;

    // ---- loop-invariant global source pointers (k=0) and LDS dest offsets (buf 0)
    const unsigned short* aSrc[4];
    unsigned aOff[4];
    #pragma unroll
    for (int i = 0; i < 4; ++i) {
        int idx = tid + i * NTHREADS;           // 0..1023
        int s   = idx >> 9;
        int rem = idx & 511;
        int row = rem >> 2;
        int q   = rem & 3;
        aSrc[i] = (s ? Hb : Xb) + (size_t)(m0 + row) * HH + q * 8;
        aOff[i] = (unsigned)(size_t)&As[0][s][row][q * 8];
    }
    const unsigned short* wSrc[6];
    unsigned wOff[6];
    #pragma unroll
    for (int i = 0; i < 6; ++i) {
        int idx = tid + i * NTHREADS;           // 0..1535
        int g   = idx >> 8;
        int rem = idx & 255;
        int row = rem >> 2;
        int q   = rem & 3;
        const unsigned short* base = (g < 3) ? (WihT + (size_t)g       * HH * HH)
                                             : (WhhT + (size_t)(g - 3) * HH * HH);
        wSrc[i] = base + (size_t)(n0 + row) * HH + q * 8;
        wOff[i] = (unsigned)(size_t)&Bs[0][g][row][q * 8];
    }

    v8f acc[6][2][2] = {};             // [gate][mi][ni], f32 16x16 each

    // ---- prologue: async-copy tile 0 into buffer 0
    #pragma unroll
    for (int i = 0; i < 4; ++i) async_copy16(aOff[i], aSrc[i]);
    #pragma unroll
    for (int i = 0; i < 6; ++i) async_copy16(wOff[i], wSrc[i]);
    wait_async0();
    __syncthreads();

    for (int kt = 1; kt <= KSTEPS; ++kt) {
        const int cb = (kt - 1) & 1;   // compute buffer
        const int lb = kt & 1;         // load buffer

        // ---- issue async DMA for tile kt into buf lb (overlaps compute below)
        if (kt < KSTEPS) {
            const int k0 = kt * BK;
            #pragma unroll
            for (int i = 0; i < 4; ++i)
                async_copy16(aOff[i] + lb * ABUF_BYTES, aSrc[i] + k0);
            #pragma unroll
            for (int i = 0; i < 6; ++i)
                async_copy16(wOff[i] + lb * WBUF_BYTES, wSrc[i] + k0);
        }

        // ---- fragments for tile kt-1, then dense WMMA block
        BFrag af[2][2];
        #pragma unroll
        for (int s = 0; s < 2; ++s)
            #pragma unroll
            for (int mi = 0; mi < 2; ++mi) {
                int row = wm * 32 + mi * 16 + lr;
                af[s][mi].d[0] = *(const v4u*)&As[cb][s][row][lh * 8];
                af[s][mi].d[1] = *(const v4u*)&As[cb][s][row][lh * 8 + 16];
            }
        BFrag bf[6][2];
        #pragma unroll
        for (int g = 0; g < 6; ++g)
            #pragma unroll
            for (int ni = 0; ni < 2; ++ni) {
                int nrow = wn * 32 + ni * 16 + lr;
                bf[g][ni].d[0] = *(const v4u*)&Bs[cb][g][nrow][lh * 8];
                bf[g][ni].d[1] = *(const v4u*)&Bs[cb][g][nrow][lh * 8 + 16];
            }
        #pragma unroll
        for (int g = 0; g < 6; ++g) {
            const int s = g / 3;                // 0: uses X, 1: uses Hx
            #pragma unroll
            for (int ni = 0; ni < 2; ++ni)
                #pragma unroll
                for (int mi = 0; mi < 2; ++mi)
                    acc[g][mi][ni] = __builtin_amdgcn_wmma_f32_16x16x32_bf16(
                        false, af[s][mi].v, false, bf[g][ni].v,
                        (short)0, acc[g][mi][ni], false, false);
        }

        // ---- make async tile visible to all waves before next compute phase
        if (kt < KSTEPS) {
            wait_async0();
            __syncthreads();
        }
    }

    // ---- epilogue: bias + gate nonlinearity + blend with hx, write fp32
    const int mgb = m0 + wm * 32;
    const int ngb = n0 + wn * 32;
    #pragma unroll
    for (int mi = 0; mi < 2; ++mi) {
        #pragma unroll
        for (int ni = 0; ni < 2; ++ni) {
            int col = ngb + ni * 16 + lr;
            float bi0 = bih[0 * HH + col], bi1 = bih[1 * HH + col], bi2 = bih[2 * HH + col];
            float bh0 = bhh[0 * HH + col], bh1 = bhh[1 * HH + col], bh2 = bhh[2 * HH + col];
            #pragma unroll
            for (int r = 0; r < 8; ++r) {
                int row = mgb + mi * 16 + r + lh * 8;
                float hv = hx[(size_t)row * HH + col];
                float gr = acc[0][mi][ni][r] + bi0 + acc[3][mi][ni][r] + bh0;
                float gz = acc[1][mi][ni][r] + bi1 + acc[4][mi][ni][r] + bh1;
                float rg = 1.0f / (1.0f + __expf(-gr));
                float zg = 1.0f / (1.0f + __expf(-gz));
                float gn = acc[2][mi][ni][r] + bi2 + rg * (acc[5][mi][ni][r] + bh2);
                float ng = tanhf(gn);
                out[(size_t)row * HH + col] = (1.0f - zg) * ng + zg * hv;
            }
        }
    }
}

// ---------------------------------------------------------------------------
extern "C" void kernel_launch(void* const* d_in, const int* in_sizes, int n_in,
                              void* d_out, int out_size, void* d_ws, size_t ws_size,
                              hipStream_t stream) {
    (void)in_sizes; (void)n_in; (void)out_size; (void)ws_size;
    const float* x   = (const float*)d_in[0];
    const float* hx  = (const float*)d_in[1];
    const float* wih = (const float*)d_in[2];
    const float* whh = (const float*)d_in[3];
    const float* bih = (const float*)d_in[4];
    const float* bhh = (const float*)d_in[5];
    float* out = (float*)d_out;

    // Workspace layout (bf16): Xb(16MB) | Hb(16MB) | WihT(24MB) | WhhT(24MB) = 80MB
    unsigned short* Xb   = (unsigned short*)d_ws;
    unsigned short* Hb   = Xb   + (size_t)BB * HH;
    unsigned short* WihT = Hb   + (size_t)BB * HH;
    unsigned short* WhhT = WihT + (size_t)3 * HH * HH;

    // 1) activations fp32 -> bf16 (2M float4's)
    cvt_act<<<dim3((BB * HH / 4) / 256), 256, 0, stream>>>(
        (const float4*)x, (const float4*)hx, Xb, Hb);

    // 2) weights fp32 [K][N] -> bf16 [N][K]
    cvt_wt<<<dim3(HH / 64, HH / 64, 6), 256, 0, stream>>>(wih, whh, WihT, WhhT);

    // 3) fused 6-GEMM + GRU combine (async-DMA double-buffered WMMA kernel)
    gru_wmma<<<dim3(HH / BN, BB / BM), NTHREADS, 0, stream>>>(
        Xb, Hb, WihT, WhhT, bih, bhh, hx, out);
}
